// TreeLstmDecoderComplete_51754355917091
// MI455X (gfx1250) — matrix operand order repaired and checked by reading, not compiled, not run on hardware
//
#include <hip/hip_runtime.h>
#include <hip/hip_bf16.h>

#define N_NODES 65536
#define N_TREES 32
#define LATENT  256
#define EMB     128
#define VOCAB   600
#define VOCAB_PAD 640
#define LEVELS  12
#define GATE4   1024
#define ROW_TILES (N_NODES / 16)

typedef __bf16 bf16_t;
typedef __attribute__((ext_vector_type(16))) __bf16 v16bf;
typedef __attribute__((ext_vector_type(8)))  __bf16 v8bf;
typedef __attribute__((ext_vector_type(4)))  __bf16 v4bf;
typedef __attribute__((ext_vector_type(8)))  float  v8f;
typedef __attribute__((ext_vector_type(4)))  float  v4f;

// ---------------------------------------------------------------------------
// WMMA helpers (CDNA5 v_wmma_f32_16x16x32_bf16, wave32)
// ---------------------------------------------------------------------------
__device__ __forceinline__ v8f wmma_bf(v16bf a, v16bf b, v8f c) {
  // (neg_a, A, neg_b, B, c_mod, C, reuse_a, reuse_b)
  return __builtin_amdgcn_wmma_f32_16x16x32_bf16(false, a, false, b, (short)0, c,
                                                 false, false);
}

// A fragment: 16x32 bf16 from row-major [16][ld] (LDS tile).
// lanes 0-15: M=lane, elements = row[k0+half*8 + {0..7, 16..23}] -> 2x b128 loads.
__device__ __forceinline__ v16bf load_a_bf(const bf16_t* A, int ld, int k0, int lane) {
  int half = lane >> 4, m = lane & 15;
  const bf16_t* row = A + (size_t)m * ld + k0 + half * 8;
  v16bf f;
#pragma unroll
  for (int e = 0; e < 8; ++e) f[e] = row[e];
#pragma unroll
  for (int e = 8; e < 16; ++e) f[e] = row[8 + e];  // row[16..23]
  return f;
}

// B fragment from pre-swizzled weights: per (ktile, ntile) 32 lanes x 16 bf16
// contiguous (32B per lane) -> one aligned 32-byte load.
__device__ __forceinline__ v16bf load_b_sw(const bf16_t* B, int ntiles, int kt,
                                           int nt, int lane) {
  return *(const v16bf*)(B + (((size_t)(kt * ntiles + nt) * 32 + lane) << 4));
}

__device__ __forceinline__ float sigm(float x) { return 1.0f / (1.0f + __expf(-x)); }
__device__ __forceinline__ float bcel(float x, float y) {
  return fmaxf(x, 0.0f) - x * y + log1pf(__expf(-fabsf(x)));
}

// ---------------------------------------------------------------------------
// Prep kernels
// ---------------------------------------------------------------------------
__global__ void k_zero_f32(float* p, int n) {
  int i = blockIdx.x * blockDim.x + threadIdx.x;
  if (i < n) p[i] = 0.0f;
}

// Swizzle a weight matrix into WMMA-B fragment-native layout.
// Logical B is [K][N] (N padded to mult of 16, zeros beyond Nsrc).
// transpose==0: B[k][n] = src[k*Nsrc + n]   (src is [K][Nsrc])
// transpose==1: B[k][n] = src[n*K + k]      (src is [Nsrc][K])
// dst element for (kt,nt,lane,e): k = kt*32 + (lane>>4)*8 + (e<8 ? e : e+8), n = nt*16 + (lane&15)
__global__ void k_swizzleB(const float* src, bf16_t* dst, int K, int N, int Nsrc,
                           int transpose) {
  int idx = blockIdx.x * blockDim.x + threadIdx.x;
  if (idx >= K * N) return;
  int e = idx & 15;
  int lane = (idx >> 4) & 31;
  int tileIdx = idx >> 9;
  int ntiles = N >> 4;
  int nt = tileIdx % ntiles;
  int kt = tileIdx / ntiles;
  int k = kt * 32 + ((lane >> 4) << 3) + (e < 8 ? e : e + 8);
  int n = nt * 16 + (lane & 15);
  float v = 0.0f;
  if (n < Nsrc)
    v = transpose ? src[(size_t)n * K + k] : src[(size_t)k * Nsrc + n];
  dst[idx] = (bf16_t)v;
}

// root_h[t][j] = b_l2h[j] + sum_k z[t][k] * W_l2h[j][k]
__global__ __launch_bounds__(256) void k_root(const float* z, const float* W,
                                              const float* b, float* root_h) {
  int t = blockIdx.x, j = threadIdx.x;
  float acc = b[j];
  const float* zr = z + (size_t)t * LATENT;
  const float* wr = W + (size_t)j * LATENT;
  for (int k = 0; k < LATENT; ++k) acc += zr[k] * wr[k];
  root_h[t * LATENT + j] = acc;
}

__global__ void k_init_state(const int* node_level, const int* tree_id,
                             const float* root_h, float* h_state, float* c_state) {
  size_t idx = (size_t)blockIdx.x * blockDim.x + threadIdx.x;
  if (idx >= (size_t)N_NODES * LATENT) return;
  int n = (int)(idx >> 8), j = (int)(idx & 255);
  h_state[idx] = (node_level[n] == 0) ? root_h[tree_id[n] * LATENT + j] : 0.0f;
  c_state[idx] = 0.0f;
}

__global__ void k_lists_init(int* lists, int* counts) {
  int i = blockIdx.x * blockDim.x + threadIdx.x;
  if (i < LEVELS * N_NODES) lists[i] = 0;
  if (i < 16) counts[i] = 0;
}

__global__ void k_build_lists(const int* node_level, int* lists, int* counts) {
  int n = blockIdx.x * blockDim.x + threadIdx.x;
  if (n >= N_NODES) return;
  int l = node_level[n];
  int pos = atomicAdd(&counts[l], 1);
  lists[l * N_NODES + pos] = n;
}

// ---------------------------------------------------------------------------
// Per-level kernel 1: snapshot parent cell state (pre-update semantics)
// ---------------------------------------------------------------------------
__global__ __launch_bounds__(256) void k_gather_cp(const int* list, const int* cnt,
                                                   const int* parent,
                                                   const float* c_state, float* Cp) {
  int tile = blockIdx.x;
  if (tile * 16 >= *cnt) return;
  for (int idx = threadIdx.x; idx < 16 * (LATENT / 4); idx += 256) {
    int m = idx >> 6, j4 = (idx & 63) << 2;
    int node = list[tile * 16 + m];
    const v4f* s = (const v4f*)&c_state[(size_t)parent[node] * LATENT + j4];
    *(v4f*)&Cp[(size_t)(tile * 16 + m) * LATENT + j4] = *s;
  }
}

// ---------------------------------------------------------------------------
// Per-level kernel 2: Hpred = tanh(Hp @ Wph1 + Hs @ Wph2 + b1 + b2)  [R,256]
// ---------------------------------------------------------------------------
__global__ __launch_bounds__(256) void k_hpred(
    const int* list, const int* cnt, const float* h_state, const int* parent,
    const int* sib, const int* hasprev, const bf16_t* Wph1s, const bf16_t* Wph2s,
    const float* b1, const float* b2, bf16_t* Hpred) {
  int tile = blockIdx.x;
  if (tile * 16 >= *cnt) return;
  __shared__ bf16_t Ahp[16][LATENT];
  __shared__ bf16_t Ahs[16][LATENT];
  int tid = threadIdx.x;
  for (int idx = tid; idx < 16 * (LATENT / 4); idx += 256) {
    int m = idx >> 6, j4 = (idx & 63) << 2;
    int node = list[tile * 16 + m];
    v4f hp = *(const v4f*)&h_state[(size_t)parent[node] * LATENT + j4];
    v4f hs = {};
    if (hasprev[node]) hs = *(const v4f*)&h_state[(size_t)sib[node] * LATENT + j4];
    v4bf hpb, hsb;
#pragma unroll
    for (int e = 0; e < 4; ++e) { hpb[e] = (bf16_t)hp[e]; hsb[e] = (bf16_t)hs[e]; }
    *(v4bf*)&Ahp[m][j4] = hpb;
    *(v4bf*)&Ahs[m][j4] = hsb;
  }
  __syncthreads();
  int wave = tid >> 5, lane = tid & 31;
  v8f acc0 = {}, acc1 = {};
  for (int k0 = 0; k0 < LATENT; k0 += 32) {
    int kt = k0 >> 5;
    v16bf a = load_a_bf(&Ahp[0][0], LATENT, k0, lane);
    acc0 = wmma_bf(a, load_b_sw(Wph1s, 16, kt, wave * 2, lane), acc0);
    acc1 = wmma_bf(a, load_b_sw(Wph1s, 16, kt, wave * 2 + 1, lane), acc1);
  }
  for (int k0 = 0; k0 < LATENT; k0 += 32) {
    int kt = k0 >> 5;
    v16bf a = load_a_bf(&Ahs[0][0], LATENT, k0, lane);
    acc0 = wmma_bf(a, load_b_sw(Wph2s, 16, kt, wave * 2, lane), acc0);
    acc1 = wmma_bf(a, load_b_sw(Wph2s, 16, kt, wave * 2 + 1, lane), acc1);
  }
  int half = lane >> 4, n = lane & 15;
#pragma unroll
  for (int r = 0; r < 8; ++r) {
    int m = r + 8 * half;
    size_t row = (size_t)(tile * 16 + m);
    int c0 = wave * 32 + n, c1 = wave * 32 + 16 + n;
    Hpred[row * LATENT + c0] = (bf16_t)tanhf(acc0[r] + b1[c0] + b2[c0]);
    Hpred[row * LATENT + c1] = (bf16_t)tanhf(acc1[r] + b1[c1] + b2[c1]);
  }
}

// ---------------------------------------------------------------------------
// Per-level kernel 3: Hpu = Hp + sigmoid(Hs @ Wgate + bg) * Hs   [R,256]
// ---------------------------------------------------------------------------
__global__ __launch_bounds__(256) void k_gate(
    const int* list, const int* cnt, const float* h_state, const int* parent,
    const int* sib, const int* hasprev, const bf16_t* Wgates, const float* bg,
    bf16_t* Hpu) {
  int tile = blockIdx.x;
  if (tile * 16 >= *cnt) return;
  __shared__ bf16_t Ahs[16][LATENT];
  int tid = threadIdx.x;
  for (int idx = tid; idx < 16 * (LATENT / 4); idx += 256) {
    int m = idx >> 6, j4 = (idx & 63) << 2;
    int node = list[tile * 16 + m];
    v4f hs = {};
    if (hasprev[node]) hs = *(const v4f*)&h_state[(size_t)sib[node] * LATENT + j4];
    v4bf hsb;
#pragma unroll
    for (int e = 0; e < 4; ++e) hsb[e] = (bf16_t)hs[e];
    *(v4bf*)&Ahs[m][j4] = hsb;
  }
  __syncthreads();
  int wave = tid >> 5, lane = tid & 31;
  v8f acc0 = {}, acc1 = {};
  for (int k0 = 0; k0 < LATENT; k0 += 32) {
    int kt = k0 >> 5;
    v16bf a = load_a_bf(&Ahs[0][0], LATENT, k0, lane);
    acc0 = wmma_bf(a, load_b_sw(Wgates, 16, kt, wave * 2, lane), acc0);
    acc1 = wmma_bf(a, load_b_sw(Wgates, 16, kt, wave * 2 + 1, lane), acc1);
  }
  int half = lane >> 4, n = lane & 15;
#pragma unroll
  for (int r = 0; r < 8; ++r) {
    int m = r + 8 * half;
    int row = tile * 16 + m;
    int node = list[row];
    int par = parent[node];
    int hp_sib = hasprev[node];
    int sb = sib[node];
#pragma unroll
    for (int half2 = 0; half2 < 2; ++half2) {
      int c = wave * 32 + half2 * 16 + n;
      float g = sigm((half2 ? acc1[r] : acc0[r]) + bg[c]);
      float hp = h_state[(size_t)par * LATENT + c];
      float hs = hp_sib ? h_state[(size_t)sb * LATENT + c] : 0.0f;
      Hpu[(size_t)row * LATENT + c] = (bf16_t)(hp + g * hs);
    }
  }
}

// ---------------------------------------------------------------------------
// Per-level kernel 4: logits = Hpred @ Wpred + b -> CE; topo -> BCE; loss add
// ---------------------------------------------------------------------------
__global__ __launch_bounds__(256) void k_loss(
    const int* list, const int* cnt, const bf16_t* Hpred, const bf16_t* Wpreds,
    const float* b_pred, const float* W_topo, const float* b_topo,
    const int* features, const int* is_parent, const int* has_sib,
    const int* is_res, float* out) {
  int tile = blockIdx.x;
  int R = *cnt;
  if (tile * 16 >= R) return;
  __shared__ bf16_t A[16][LATENT];        // 8 KB
  __shared__ float logits[16][VOCAB_PAD]; // 40 KB
  int tid = threadIdx.x;
  for (int idx = tid; idx < 16 * (LATENT / 8); idx += 256) {
    int m = idx >> 5, j8 = (idx & 31) << 3;
    *(v8bf*)&A[m][j8] =
        *(const v8bf*)&Hpred[(size_t)(tile * 16 + m) * LATENT + j8];
  }
  __syncthreads();
  int wave = tid >> 5, lane = tid & 31;
  for (int t = wave; t < VOCAB_PAD / 16; t += 8) {
    v8f acc = {};
    for (int k0 = 0; k0 < LATENT; k0 += 32) {
      v16bf a = load_a_bf(&A[0][0], LATENT, k0, lane);
      acc = wmma_bf(a, load_b_sw(Wpreds, VOCAB_PAD / 16, k0 >> 5, t, lane), acc);
    }
    int half = lane >> 4, n = lane & 15, col = t * 16 + n;
#pragma unroll
    for (int r = 0; r < 8; ++r)
      logits[r + 8 * half][col] = acc[r] + (col < VOCAB ? b_pred[col] : -3.0e38f);
  }
  __syncthreads();
  // per-row reductions: 16 threads per row (xor masks stay within 16-lane halves)
  int m = tid >> 4, lg = tid & 15;
  float mx = -3.0e38f;
  for (int j = lg; j < VOCAB; j += 16) mx = fmaxf(mx, logits[m][j]);
#pragma unroll
  for (int s = 8; s; s >>= 1) mx = fmaxf(mx, __shfl_xor(mx, s, 32));
  float se = 0.0f;
  for (int j = lg; j < VOCAB; j += 16) se += __expf(logits[m][j] - mx);
#pragma unroll
  for (int s = 8; s; s >>= 1) se += __shfl_xor(se, s, 32);
  float logZ = mx + __logf(se);
  float d0 = 0.f, d1 = 0.f, d2 = 0.f;
  for (int k = lg; k < LATENT; k += 16) {
    float hv = (float)A[m][k];
    d0 += hv * W_topo[k * 3 + 0];
    d1 += hv * W_topo[k * 3 + 1];
    d2 += hv * W_topo[k * 3 + 2];
  }
#pragma unroll
  for (int s = 8; s; s >>= 1) {
    d0 += __shfl_xor(d0, s, 32);
    d1 += __shfl_xor(d1, s, 32);
    d2 += __shfl_xor(d2, s, 32);
  }
  int row = tile * 16 + m;
  if (lg == 0 && row < R) {
    int node = list[row];
    float ce = logZ - logits[m][features[node]];
    float bce = bcel(d0 + b_topo[0], (float)is_parent[node]) +
                bcel(d1 + b_topo[1], (float)has_sib[node]) +
                bcel(d2 + b_topo[2], (float)is_res[node]);
    atomicAdd(out, (ce + bce) * (1.0f / (float)N_NODES));
  }
}

// ---------------------------------------------------------------------------
// Per-level kernel 5: LSTM cell gates via WMMA; each wave owns i/f/g/o tiles
// for its 16 hidden units -> full cell update in registers -> scatter h/c.
// ---------------------------------------------------------------------------
__global__ __launch_bounds__(256) void k_lstm(
    const int* list, const int* cnt, const float* emb_table, const int* features,
    const bf16_t* Hpu, const bf16_t* Wihs, const bf16_t* Whhs, const float* b_ih,
    const float* b_hh, const float* Cp, float* h_state, float* c_state) {
  int tile = blockIdx.x;
  int R = *cnt;
  if (tile * 16 >= R) return;
  __shared__ bf16_t A[16][EMB + LATENT]; // 12 KB: [emb | h_par_upd]
  int tid = threadIdx.x;
  for (int idx = tid; idx < 16 * ((EMB + LATENT) / 4); idx += 256) {
    int m = idx / 96, j4 = (idx % 96) << 2;
    int node = list[tile * 16 + m];
    if (j4 < EMB) {
      v4f ev = *(const v4f*)&emb_table[(size_t)features[node] * EMB + j4];
      v4bf eb;
#pragma unroll
      for (int e = 0; e < 4; ++e) eb[e] = (bf16_t)ev[e];
      *(v4bf*)&A[m][j4] = eb;
    } else {
      *(v4bf*)&A[m][j4] =
          *(const v4bf*)&Hpu[(size_t)(tile * 16 + m) * LATENT + (j4 - EMB)];
    }
  }
  __syncthreads();
  int wave = tid >> 5, lane = tid & 31;
  for (int t = wave; t < LATENT / 16; t += 8) { // 16 unit tiles / 8 waves
    v8f ai = {}, af = {}, ag = {}, ao = {};
    for (int k0 = 0; k0 < EMB + LATENT; k0 += 32) {
      v16bf a = load_a_bf(&A[0][0], EMB + LATENT, k0, lane);
      const bf16_t* B = (k0 < EMB) ? Wihs : Whhs;
      int kt = ((k0 < EMB) ? k0 : (k0 - EMB)) >> 5;
      ai = wmma_bf(a, load_b_sw(B, 64, kt, 0 * 16 + t, lane), ai);
      af = wmma_bf(a, load_b_sw(B, 64, kt, 1 * 16 + t, lane), af);
      ag = wmma_bf(a, load_b_sw(B, 64, kt, 2 * 16 + t, lane), ag);
      ao = wmma_bf(a, load_b_sw(B, 64, kt, 3 * 16 + t, lane), ao);
    }
    int half = lane >> 4, n = lane & 15, j = t * 16 + n;
    float bI = b_ih[j] + b_hh[j];
    float bF = b_ih[LATENT + j] + b_hh[LATENT + j];
    float bG = b_ih[2 * LATENT + j] + b_hh[2 * LATENT + j];
    float bO = b_ih[3 * LATENT + j] + b_hh[3 * LATENT + j];
#pragma unroll
    for (int r = 0; r < 8; ++r) {
      int m = r + 8 * half;
      int row = tile * 16 + m;
      if (row < R) {
        int node = list[row];
        float cp = Cp[(size_t)row * LATENT + j];
        float cn = sigm(af[r] + bF) * cp + sigm(ai[r] + bI) * tanhf(ag[r] + bG);
        float hn = sigm(ao[r] + bO) * tanhf(cn);
        c_state[(size_t)node * LATENT + j] = cn;
        h_state[(size_t)node * LATENT + j] = hn;
      }
    }
  }
}

// ---------------------------------------------------------------------------
// Host launcher
// ---------------------------------------------------------------------------
extern "C" void kernel_launch(void* const* d_in, const int* in_sizes, int n_in,
                              void* d_out, int out_size, void* d_ws, size_t ws_size,
                              hipStream_t stream) {
  const float* z         = (const float*)d_in[0];
  const float* emb_table = (const float*)d_in[1];
  const float* W_l2h     = (const float*)d_in[2];
  const float* b_l2h     = (const float*)d_in[3];
  const float* W_ih      = (const float*)d_in[4];
  const float* W_hh      = (const float*)d_in[5];
  const float* b_ih      = (const float*)d_in[6];
  const float* b_hh      = (const float*)d_in[7];
  const float* W_ph1     = (const float*)d_in[8];
  const float* b_ph1     = (const float*)d_in[9];
  const float* W_ph2     = (const float*)d_in[10];
  const float* b_ph2     = (const float*)d_in[11];
  const float* W_gate    = (const float*)d_in[12];
  const float* b_gate    = (const float*)d_in[13];
  const float* W_topo    = (const float*)d_in[14];
  const float* b_topo    = (const float*)d_in[15];
  const float* W_pred    = (const float*)d_in[16];
  const float* b_pred    = (const float*)d_in[17];
  const int* features    = (const int*)d_in[18];
  const int* node_level  = (const int*)d_in[19];
  const int* parent_idx  = (const int*)d_in[20];
  const int* sibling_idx = (const int*)d_in[21];
  const int* has_prev    = (const int*)d_in[22];
  const int* tree_id     = (const int*)d_in[23];
  const int* is_parent   = (const int*)d_in[24];
  const int* has_sib_lbl = (const int*)d_in[25];
  const int* is_res      = (const int*)d_in[26];
  float* out = (float*)d_out;

  char* ws = (char*)d_ws;
  size_t off = 0;
  auto alloc = [&](size_t bytes) -> void* {
    void* p = ws + off;
    off = (off + bytes + 255) & ~(size_t)255;
    return p;
  };
  const size_t NP = N_NODES + 16;
  float*  h_state = (float*)alloc((size_t)N_NODES * LATENT * 4);
  float*  c_state = (float*)alloc((size_t)N_NODES * LATENT * 4);
  float*  Cp      = (float*)alloc(NP * LATENT * 4);
  bf16_t* Hpred   = (bf16_t*)alloc(NP * LATENT * 2);
  bf16_t* Hpu     = (bf16_t*)alloc(NP * LATENT * 2);
  bf16_t* Wph1s   = (bf16_t*)alloc((size_t)LATENT * LATENT * 2);
  bf16_t* Wph2s   = (bf16_t*)alloc((size_t)LATENT * LATENT * 2);
  bf16_t* Wgates  = (bf16_t*)alloc((size_t)LATENT * LATENT * 2);
  bf16_t* Wpreds  = (bf16_t*)alloc((size_t)LATENT * VOCAB_PAD * 2);
  bf16_t* Wihs    = (bf16_t*)alloc((size_t)EMB * GATE4 * 2);
  bf16_t* Whhs    = (bf16_t*)alloc((size_t)LATENT * GATE4 * 2);
  float*  root_h  = (float*)alloc((size_t)N_TREES * LATENT * 4);
  int*    counts  = (int*)alloc(64);
  int*    lists   = (int*)alloc((size_t)LEVELS * N_NODES * 4);
  (void)ws_size; (void)n_in; (void)in_sizes;

  // --- one-time prep -------------------------------------------------------
  k_zero_f32<<<(out_size + 255) / 256, 256, 0, stream>>>(out, out_size);
  k_swizzleB<<<(LATENT * LATENT + 255) / 256, 256, 0, stream>>>(
      W_ph1, Wph1s, LATENT, LATENT, LATENT, 0);
  k_swizzleB<<<(LATENT * LATENT + 255) / 256, 256, 0, stream>>>(
      W_ph2, Wph2s, LATENT, LATENT, LATENT, 0);
  k_swizzleB<<<(LATENT * LATENT + 255) / 256, 256, 0, stream>>>(
      W_gate, Wgates, LATENT, LATENT, LATENT, 0);
  k_swizzleB<<<(LATENT * VOCAB_PAD + 255) / 256, 256, 0, stream>>>(
      W_pred, Wpreds, LATENT, VOCAB_PAD, VOCAB, 0);
  k_swizzleB<<<(EMB * GATE4 + 255) / 256, 256, 0, stream>>>(
      W_ih, Wihs, EMB, GATE4, GATE4, 1);
  k_swizzleB<<<(LATENT * GATE4 + 255) / 256, 256, 0, stream>>>(
      W_hh, Whhs, LATENT, GATE4, GATE4, 1);
  k_root<<<N_TREES, 256, 0, stream>>>(z, W_l2h, b_l2h, root_h);
  k_init_state<<<(N_NODES * LATENT) / 256, 256, 0, stream>>>(node_level, tree_id,
                                                             root_h, h_state, c_state);
  k_lists_init<<<(LEVELS * N_NODES + 255) / 256, 256, 0, stream>>>(lists, counts);
  k_build_lists<<<N_NODES / 256, 256, 0, stream>>>(node_level, lists, counts);

  // --- per-level pipeline --------------------------------------------------
  for (int lvl = 1; lvl < LEVELS; ++lvl) {
    const int* list = lists + (size_t)lvl * N_NODES;
    const int* cnt  = counts + lvl;
    k_gather_cp<<<ROW_TILES, 256, 0, stream>>>(list, cnt, parent_idx, c_state, Cp);
    k_hpred<<<ROW_TILES, 256, 0, stream>>>(list, cnt, h_state, parent_idx,
                                           sibling_idx, has_prev, Wph1s, Wph2s,
                                           b_ph1, b_ph2, Hpred);
    k_gate<<<ROW_TILES, 256, 0, stream>>>(list, cnt, h_state, parent_idx,
                                          sibling_idx, has_prev, Wgates, b_gate, Hpu);
    k_loss<<<ROW_TILES, 256, 0, stream>>>(list, cnt, Hpred, Wpreds, b_pred, W_topo,
                                          b_topo, features, is_parent, has_sib_lbl,
                                          is_res, out);
    k_lstm<<<ROW_TILES, 256, 0, stream>>>(list, cnt, emb_table, features, Hpu,
                                          Wihs, Whhs, b_ih, b_hh, Cp, h_state,
                                          c_state);
  }
}
